// UCBAttention_60481729462842
// MI455X (gfx1250) — matrix operand order repaired and checked by reading, not compiled, not run on hardware
//
#include <hip/hip_runtime.h>
#include <cmath>

// ---- problem constants -----------------------------------------------------
#define Hh   12
#define Bb   16
#define Nn   577
#define Cc   768
#define HDim 64
#define NP   608          /* 577 padded to 19*32 (multiple of the K=32 WMMA step) */
#define NT   (Bb*Nn)      /* 9232 = 577*16: token rows tile exactly by 16 */
#define RB   37           /* row-blocks of 16 covering 577 queries */
#define TOPK 288
#define W3C  (3*Cc)       /* 2304 */

typedef __bf16 bf16;
typedef __attribute__((ext_vector_type(16))) __bf16 v16bf;
typedef __attribute__((ext_vector_type(8)))  __bf16 v8bf;
typedef __attribute__((ext_vector_type(8)))  float  v8f;

#if __has_builtin(__builtin_amdgcn_tensor_load_to_lds)
#define HAVE_TDM 1
typedef unsigned int u32x4 __attribute__((ext_vector_type(4)));
typedef int          i32x4 __attribute__((ext_vector_type(4)));
typedef int          i32x8 __attribute__((ext_vector_type(8)));
#else
#define HAVE_TDM 0
#endif

// ---- WMMA fragment helpers (CDNA5 16-bit layouts, wave32) ------------------
// A (16x32): lane m = lane&15; g = lane>>4; per-lane elements cover
// K = k0 + [8g,8g+8) then k0 + [16+8g, 16+8g+8)  -> two 16B loads.
__device__ __forceinline__ v16bf a_frag(const bf16* row, int k0, int g) {
  v8bf lo = *(const v8bf*)(row + k0 + 8 * g);
  v8bf hi = *(const v8bf*)(row + k0 + 16 + 8 * g);
  v16bf a;
#pragma unroll
  for (int e = 0; e < 8; ++e) { a[e] = lo[e]; a[e + 8] = hi[e]; }
  return a;
}
// B (32x16): lane n = lane&15; per-lane elements cover
// K = k0 + [8g,8g+8) then k0 + [16+8g, 16+8g+8)  -> two 16B loads.
__device__ __forceinline__ v16bf b_frag(const bf16* row, int k0, int g) {
  v8bf lo = *(const v8bf*)(row + k0 + 8 * g);
  v8bf hi = *(const v8bf*)(row + k0 + 16 + 8 * g);
  v16bf b;
#pragma unroll
  for (int e = 0; e < 8; ++e) { b[e] = lo[e]; b[e + 8] = hi[e]; }
  return b;
}
__device__ __forceinline__ v8f wmma_bf16(v16bf a, v16bf b, v8f c) {
  return __builtin_amdgcn_wmma_f32_16x16x32_bf16(false, a, false, b,
                                                 (short)0, c, false, false);
}
// wave32 tree reductions (uniform control flow -> EXEC stays all-ones)
__device__ __forceinline__ float wave_max(float v) {
#pragma unroll
  for (int o = 16; o > 0; o >>= 1) v = fmaxf(v, __shfl_xor(v, o, 32));
  return v;
}
__device__ __forceinline__ float wave_sum(float v) {
#pragma unroll
  for (int o = 16; o > 0; o >>= 1) v += __shfl_xor(v, o, 32);
  return v;
}

// ---- small utility kernels -------------------------------------------------
__global__ void k_cvt(const float* __restrict__ s, bf16* __restrict__ d, int n) {
  int i = blockIdx.x * blockDim.x + threadIdx.x;
  if (i < n) d[i] = (bf16)s[i];
}
__global__ void k_zero_bf16(bf16* __restrict__ d, size_t n) {
  size_t i = (size_t)blockIdx.x * blockDim.x + threadIdx.x;
  size_t stride = (size_t)gridDim.x * blockDim.x;
  for (; i < n; i += stride) d[i] = (bf16)0.f;
}

// ---- pass 1: fused QKV GEMM (bf16 WMMA), scatter into q/k/vT layouts -------
__global__ __launch_bounds__(32) void k_qkv(const bf16* __restrict__ xb,
                                            const bf16* __restrict__ wqb,
                                            bf16* __restrict__ qb,
                                            bf16* __restrict__ kb,
                                            bf16* __restrict__ vtb) {
  const int bx = blockIdx.x;            // 577 token-blocks * 144 w-tiles
  const int wt = bx % (W3C / 16);
  const int tb = bx / (W3C / 16);
  const int lane = threadIdx.x, g = lane >> 4, ln = lane & 15;
  const bf16* arow = xb  + (size_t)(tb * 16 + ln) * Cc;
  const bf16* brow = wqb + (size_t)(wt * 16 + ln) * Cc;
  v8f acc = {};
  for (int kt = 0; kt < Cc / 32; ++kt) {
    __builtin_prefetch(arow + (kt + 1) * 32, 0, 3);   // WGP-scope prefetch
    acc = wmma_bf16(a_frag(arow, kt * 32, g), b_frag(brow, kt * 32, g), acc);
  }
  const int w = wt * 16 + ln;
  const int which = w / Cc;             // uniform per block (768 % 16 == 0)
  const int h = (w % Cc) / HDim, d = w % HDim;
#pragma unroll
  for (int i = 0; i < 8; ++i) {
    const int t = tb * 16 + i + 8 * g;
    const int b = t / Nn, n = t % Nn;
    const float val = acc[i];
    const size_t bh = (size_t)(b * Hh + h);
    if (which == 0)      qb[(bh * NP + n) * HDim + d] = (bf16)(val * 0.125f); // hd^-0.5 folded in
    else if (which == 1) kb[(bh * NP + n) * HDim + d] = (bf16)val;
    else                 vtb[(bh * HDim + d) * NP + n] = (bf16)val;           // V stored transposed
  }
}

// ---- pass A: scores + softmax, per-key column means ------------------------
__global__ __launch_bounds__(32) void k_attn_colsum(const bf16* __restrict__ qb,
                                                    const bf16* __restrict__ kb,
                                                    float* __restrict__ pcol) {
  __shared__ float S[16][NP];
  __shared__ float rowinv[16];
  const int bx = blockIdx.x;            // B*H*RB
  const int rb = bx % RB, bh = bx / RB;
  const int lane = threadIdx.x, g = lane >> 4, ln = lane & 15;
  const bf16* arow  = qb + ((size_t)bh * NP + rb * 16 + ln) * HDim;
  const bf16* kbase = kb + (size_t)bh * NP * HDim;
  for (int ct = 0; ct < NP / 16; ++ct) {
    const bf16* brow = kbase + (size_t)(ct * 16 + ln) * HDim;
    v8f acc = {};
    acc = wmma_bf16(a_frag(arow, 0, g),  b_frag(brow, 0, g),  acc);
    acc = wmma_bf16(a_frag(arow, 32, g), b_frag(brow, 32, g), acc);
#pragma unroll
    for (int i = 0; i < 8; ++i) S[i + 8 * g][ct * 16 + ln] = acc[i];
  }
  __syncthreads();
  // serial over 16 rows, wave-parallel over columns
  for (int r = 0; r < 16; ++r) {
    const int n = rb * 16 + r;
    if (n < Nn) {
      float m = -3.4e38f;
      for (int j = lane; j < Nn; j += 32) m = fmaxf(m, S[r][j]);
      m = wave_max(m);
      float z = 0.f;
      for (int j = lane; j < Nn; j += 32) { float e = __expf(S[r][j] - m); S[r][j] = e; z += e; }
      z = wave_sum(z);
      if (lane == 0) rowinv[r] = 1.f / z;
    } else if (lane == 0) {
      rowinv[r] = 0.f;                  // padded query rows contribute nothing
    }
  }
  __syncthreads();
  for (int j = 1 + lane; j < Nn; j += 32) {
    float cs = 0.f;
#pragma unroll
    for (int r = 0; r < 16; ++r) cs += S[r][j] * rowinv[r];
    pcol[((size_t)bh * RB + rb) * 576 + (j - 1)] = cs * (1.f / (float)Nn);
  }
}

// ---- pass B: UCB scores, deterministic top-k rank selection ----------------
__global__ void k_ucb(const float* __restrict__ pcol, const float* __restrict__ ucbc,
                      const int* __restrict__ counterp, float* __restrict__ gub) {
  int idx = blockIdx.x * blockDim.x + threadIdx.x;
  if (idx >= Bb * 576) return;
  const int b = idx / 576, p = idx % 576;
  const float logc = logf((float)(*counterp) + 1.0f);
  float s = 0.f;
  for (int h = 0; h < Hh; ++h) {
    float ps = 0.f;
    for (int rb = 0; rb < RB; ++rb)
      ps += pcol[(((size_t)(b * Hh + h)) * RB + rb) * 576 + p];
    s += ps + sqrtf(logc / (ucbc[h * Nn + p + 1] + 1e-6f));
  }
  gub[idx] = s / (float)Hh;
}

__global__ __launch_bounds__(576) void k_topk(const float* __restrict__ gub,
                                              float* __restrict__ keep) {
  __shared__ float sc[576];
  const int b = blockIdx.x, j = threadIdx.x;
  sc[j] = gub[b * 576 + j];
  __syncthreads();
  const float mine = sc[j];
  int rank = 0;
  for (int q = 0; q < 576; ++q) {
    const float v = sc[q];
    rank += (v > mine) || (v == mine && q < j);   // stable: matches top_k tiebreak
  }
  keep[b * Nn + j + 1] = (rank < TOPK) ? 1.f : 0.f;
  if (j == 0) keep[b * Nn] = 1.f;                 // CLS always kept
}

// ---- pass C: masked softmax recompute + context GEMM (P x V) ---------------
__global__ __launch_bounds__(32) void k_attn_ctx(const bf16* __restrict__ qb,
                                                 const bf16* __restrict__ kb,
                                                 const bf16* __restrict__ vtb,
                                                 const float* __restrict__ keep,
                                                 bf16* __restrict__ ctxb) {
  __shared__ float S[16][NP];
  __shared__ bf16  P[16][NP];
  __shared__ float keepL[Nn];
  const int bx = blockIdx.x;            // B*H*RB
  const int rb = bx % RB, bh = bx / RB;
  const int b = bh / Hh, h = bh % Hh;
  const int lane = threadIdx.x, g = lane >> 4, ln = lane & 15;
  const float* keepb = keep + b * Nn;

  // Stage the keep-mask row into LDS via the Tensor Data Mover; the DMA
  // overlaps the 76-WMMA score phase below, waited on before the masked phase.
#if HAVE_TDM
  {
    const unsigned lds_off = (unsigned)(size_t)&keepL[0];
    const unsigned long long ga = (unsigned long long)(size_t)keepb;
    u32x4 g0;
    g0[0] = 1u;                                               // count=1 (valid user D#)
    g0[1] = lds_off;                                          // lds_addr
    g0[2] = (unsigned)ga;                                     // global_addr[31:0]
    g0[3] = (unsigned)((ga >> 32) & 0x1FFFFFFu) | (2u << 30); // global_addr[56:32] | type=2
    i32x8 g1;
    g1[0] = (2 << 16);                                        // data_size = 4 bytes
    g1[1] = (Nn & 0xFFFF) << 16;                              // tensor_dim0[15:0]
    g1[2] = (1 << 16);                                        // tensor_dim1 = 1
    g1[3] = (Nn << 16);                                       // tile_dim0 = 577
    g1[4] = 1;                                                // tile_dim1 = 1
    g1[5] = Nn;                                               // tensor_dim0_stride
    g1[6] = 0; g1[7] = 0;
    i32x4 g2 = {0, 0, 0, 0}, g3 = {0, 0, 0, 0};
    i32x8 g4 = {0, 0, 0, 0, 0, 0, 0, 0};                      // clang-23 6-arg form
    __builtin_amdgcn_tensor_load_to_lds(g0, g1, g2, g3, g4, 0);
  }
#else
  for (int j = threadIdx.x; j < Nn; j += 32) keepL[j] = keepb[j];
#endif

  const bf16* arow  = qb + ((size_t)bh * NP + rb * 16 + ln) * HDim;
  const bf16* kbase = kb + (size_t)bh * NP * HDim;
  for (int ct = 0; ct < NP / 16; ++ct) {
    const bf16* brow = kbase + (size_t)(ct * 16 + ln) * HDim;
    v8f acc = {};
    acc = wmma_bf16(a_frag(arow, 0, g),  b_frag(brow, 0, g),  acc);
    acc = wmma_bf16(a_frag(arow, 32, g), b_frag(brow, 32, g), acc);
#pragma unroll
    for (int i = 0; i < 8; ++i) S[i + 8 * g][ct * 16 + ln] = acc[i];
  }
#if HAVE_TDM
  __builtin_amdgcn_s_wait_tensorcnt(0);
#endif
  __syncthreads();

  // serial over rows, wave-parallel over columns; uniform branches only
  for (int r = 0; r < 16; ++r) {
    const int n = rb * 16 + r;
    if (n < Nn) {
      float m = -3.4e38f;
      for (int j = lane; j < Nn; j += 32) m = fmaxf(m, S[r][j]);
      m = wave_max(m);
      float z = 0.f;
      for (int j = lane; j < Nn; j += 32) { float e = __expf(S[r][j] - m); S[r][j] = e; z += e; }
      z = wave_sum(z);
      float kd = 0.f;
      for (int j = lane; j < Nn; j += 32) kd += S[r][j] * keepL[j];
      kd = wave_sum(kd);
      const float rowkeep = keepL[n];
      // pruned = (e/z * mask) / (sum(e/z * mask) + 1e-8)  ==  e*mask / D
      const float D = ((rowkeep > 0.5f) ? z : kd) + 1e-8f * z;
      const float invD = 1.f / D;
      if (rowkeep > 0.5f) {
        for (int j = lane; j < Nn; j += 32) P[r][j] = (bf16)(S[r][j] * invD);
      } else {
        for (int j = lane; j < Nn; j += 32) P[r][j] = (bf16)(S[r][j] * keepL[j] * invD);
      }
      for (int j = Nn + lane; j < NP; j += 32) P[r][j] = (bf16)0.f;
    } else {
      for (int j = lane; j < NP; j += 32) P[r][j] = (bf16)0.f;
    }
  }
  __syncthreads();

  const bf16* vbase = vtb + (size_t)bh * HDim * NP;
  v8f acc[4] = {};
  for (int kt = 0; kt < NP / 32; ++kt) {
    v16bf a = a_frag(&P[ln][0], kt * 32, g);      // LDS-sourced A fragment
#pragma unroll
    for (int dt = 0; dt < 4; ++dt) {
      v16bf bf = b_frag(vbase + (size_t)(dt * 16 + ln) * NP, kt * 32, g);
      acc[dt] = wmma_bf16(a, bf, acc[dt]);
    }
  }
#pragma unroll
  for (int i = 0; i < 8; ++i) {
    const int n = rb * 16 + i + 8 * g;
    if (n < Nn) {
      const size_t trow = (size_t)(b * Nn + n) * Cc + h * HDim;
#pragma unroll
      for (int dt = 0; dt < 4; ++dt)
        ctxb[trow + dt * 16 + ln] = (bf16)acc[dt][i];
    }
  }
}

// ---- pass D: output projection GEMM + bias ---------------------------------
__global__ __launch_bounds__(32) void k_proj(const bf16* __restrict__ ctxb,
                                             const bf16* __restrict__ wpb,
                                             const float* __restrict__ bproj,
                                             float* __restrict__ out) {
  const int bx = blockIdx.x;            // 577 * 48
  const int ct = bx % (Cc / 16);
  const int tb = bx / (Cc / 16);
  const int lane = threadIdx.x, g = lane >> 4, ln = lane & 15;
  const bf16* arow = ctxb + (size_t)(tb * 16 + ln) * Cc;
  const bf16* brow = wpb  + (size_t)(ct * 16 + ln) * Cc;
  v8f acc = {};
  for (int kt = 0; kt < Cc / 32; ++kt) {
    __builtin_prefetch(arow + (kt + 1) * 32, 0, 3);   // WGP-scope prefetch
    acc = wmma_bf16(a_frag(arow, kt * 32, g), b_frag(brow, kt * 32, g), acc);
  }
  const int col = ct * 16 + ln;
  const float bias = bproj[col];
#pragma unroll
  for (int i = 0; i < 8; ++i)
    out[(size_t)(tb * 16 + i + 8 * g) * Cc + col] = acc[i] + bias;
}

// ---- score_delta = broadcast counts(from keep mask) over heads -------------
__global__ void k_sdelta(const float* __restrict__ keep, float* __restrict__ sd) {
  int i = blockIdx.x * blockDim.x + threadIdx.x;
  if (i >= Hh * Nn) return;
  const int n = i % Nn;
  float c = 0.f;
  if (n != 0) for (int b = 0; b < Bb; ++b) c += keep[b * Nn + n];
  sd[i] = c / (float)Bb;
}

// ---- host-side orchestration -----------------------------------------------
extern "C" void kernel_launch(void* const* d_in, const int* in_sizes, int n_in,
                              void* d_out, int out_size, void* d_ws, size_t ws_size,
                              hipStream_t stream) {
  (void)in_sizes; (void)n_in; (void)out_size; (void)ws_size;
  const float* x     = (const float*)d_in[0];
  const float* Wqkv  = (const float*)d_in[1];
  const float* Wproj = (const float*)d_in[2];
  const float* bproj = (const float*)d_in[3];
  const float* ucbc  = (const float*)d_in[4];
  const int*   ctr   = (const int*)d_in[5];
  // d_in[6] (ucb_enabled): output size implies the UCB branch is taken.

  char* ws = (char*)d_ws;
  size_t off = 0;
  auto alloc = [&](size_t bytes) -> char* {
    char* p = ws + off; off += (bytes + 255) & ~(size_t)255; return p;
  };
  bf16*  xb   = (bf16*)alloc((size_t)NT * Cc * 2);
  bf16*  wqb  = (bf16*)alloc((size_t)W3C * Cc * 2);
  bf16*  wpb  = (bf16*)alloc((size_t)Cc * Cc * 2);
  bf16*  qb   = (bf16*)alloc((size_t)Bb * Hh * NP * HDim * 2);
  bf16*  kb   = (bf16*)alloc((size_t)Bb * Hh * NP * HDim * 2);
  bf16*  vtb  = (bf16*)alloc((size_t)Bb * Hh * HDim * NP * 2);
  bf16*  ctxb = (bf16*)alloc((size_t)NT * Cc * 2);
  float* pcol = (float*)alloc((size_t)Bb * Hh * RB * 576 * 4);
  float* gub  = (float*)alloc((size_t)Bb * 576 * 4);
  float* keep = (float*)alloc((size_t)Bb * Nn * 4);
  float* outp = (float*)d_out;
  float* sd   = outp + (size_t)NT * Cc;

  { int n = NT * Cc;  k_cvt<<<(n + 255) / 256, 256, 0, stream>>>(x, xb, n); }
  { int n = W3C * Cc; k_cvt<<<(n + 255) / 256, 256, 0, stream>>>(Wqkv, wqb, n); }
  { int n = Cc * Cc;  k_cvt<<<(n + 255) / 256, 256, 0, stream>>>(Wproj, wpb, n); }
  // q/k/vT are laid out contiguously (each size is 256B-aligned): zero pads once.
  k_zero_bf16<<<4096, 256, 0, stream>>>(qb, (size_t)3 * Bb * Hh * NP * HDim);

  k_qkv        <<<577 * (W3C / 16), 32, 0, stream>>>(xb, wqb, qb, kb, vtb);
  k_attn_colsum<<<Bb * Hh * RB,     32, 0, stream>>>(qb, kb, pcol);
  k_ucb        <<<(Bb * 576 + 255) / 256, 256, 0, stream>>>(pcol, ucbc, ctr, gub);
  k_topk       <<<Bb, 576, 0, stream>>>(gub, keep);
  k_attn_ctx   <<<Bb * Hh * RB,     32, 0, stream>>>(qb, kb, vtb, keep, ctxb);
  k_proj       <<<577 * (Cc / 16),  32, 0, stream>>>(ctxb, wpb, bproj, outp);
  k_sdelta     <<<(Hh * Nn + 255) / 256, 256, 0, stream>>>(keep, sd);
}
